// ShowAndTell_36094905155876
// MI455X (gfx1250) — matrix operand order repaired and checked by reading, not compiled, not run on hardware
//
#include <hip/hip_runtime.h>
#include <hip/hip_bf16.h>
#include <math.h>
#include <stdint.h>

typedef __attribute__((ext_vector_type(16))) __bf16 v16bf;
typedef __attribute__((ext_vector_type(8)))  float  v8f;
typedef int vint4 __attribute__((vector_size(16)));

#ifndef __has_builtin
#define __has_builtin(x) 0
#endif
#if __has_builtin(__builtin_amdgcn_global_load_async_to_lds_b128)
#define HAVE_ASYNC_LDS 1
#else
#define HAVE_ASYNC_LDS 0
#endif
#if __has_builtin(__builtin_amdgcn_s_wait_asynccnt)
#define HAVE_WAIT_ASYNC 1
#else
#define HAVE_WAIT_ASYNC 0
#endif

namespace {
constexpr int kB      = 64;
constexpr int kSteps  = 31;          // T-1 decode steps
constexpr int kEmb    = 300;
constexpr int kEmbP   = 320;         // padded to 32
constexpr int kKtEmb  = kEmbP / 32;  // 10
constexpr int kFeat   = 4100;
constexpr int kFeatP  = 4128;
constexpr int kKtFeat = kFeatP / 32; // 129
constexpr int kHid    = 512;
constexpr int kKtHid  = kHid / 32;   // 16
constexpr int kG3     = 3 * kHid;    // 1536
constexpr int kVocab  = 32000;
constexpr int kMRows  = kSteps * kB; // 1984 (row r = t*64 + b)
constexpr int kMTiles = kMRows / 16; // 124
constexpr int kNTCls  = kVocab / 16; // 2000
}

// ---- WMMA fragment index helpers (CDNA5 ISA 7.12.2, wave32) --------------
// A (16x32, 16-bit): lane holds M=lane&15; element e -> k within 32-block:
__device__ __forceinline__ int a_k32(int lane, int e) {
  int g = e >> 1;
  return ((g >> 2) << 4) | (((lane >> 4) & 1) << 3) | ((g & 3) << 1) | (e & 1);
}
// inverse: (m_loc 0..15, k32 0..31) -> (lane, element)
__device__ __forceinline__ void a_inv(int m_loc, int k32, int& lane, int& e) {
  int half = (k32 >> 3) & 1;
  int g = ((k32 >> 4) << 2) | ((k32 & 7) >> 1);
  e = (g << 1) | (k32 & 1);
  lane = m_loc | (half << 4);
}
// B (32x16, 16-bit): lane holds N=lane&15; element e -> k = 16*(e>>3)+8*(lane>>4)+(e&7).

__device__ __forceinline__ v8f wmma_bf16(v16bf a, v16bf b, v8f c) {
  return __builtin_amdgcn_wmma_f32_16x16x32_bf16(false, a, false, b,
                                                 (short)0, c, false, false);
}

__device__ __forceinline__ float sigmoidf_(float x) {
  return 1.0f / (1.0f + __expf(-x));
}

// ---- global -> LDS 16-byte copy (async when toolchain exposes it) --------
__device__ __forceinline__ void copy16_to_lds(const __bf16* g, __bf16* l) {
#if HAVE_ASYNC_LDS
  // builtin signature (from probe): (v4i AS1*, v4i AS3*, imm offset, imm cpol)
  // generic LDS addresses carry the LDS offset in the low 32 bits
  __builtin_amdgcn_global_load_async_to_lds_b128(
      (__attribute__((address_space(1))) vint4*)(uintptr_t)g,
      (__attribute__((address_space(3))) vint4*)(uint32_t)(uintptr_t)l,
      0, 0);
#else
  *(float4*)l = *(const float4*)g;
#endif
}

__device__ __forceinline__ void stage_fence() {
#if HAVE_ASYNC_LDS
#if HAVE_WAIT_ASYNC
  __builtin_amdgcn_s_wait_asynccnt(0);
#else
  asm volatile("s_wait_asynccnt 0" ::: "memory");
#endif
#endif
  __syncthreads();
}

// ---- packing kernels -----------------------------------------------------
// dst layout: [tile][kt][lane(32)][e(16)] bf16, 32B per (tile,kt,lane).

__global__ void pack_A_feats(__bf16* __restrict__ dst,
                             const float* __restrict__ vis,
                             const float* __restrict__ tgt) {
  int idx = blockIdx.x * blockDim.x + threadIdx.x;
  const int total = 4 * kKtFeat * 32 * 16;
  if (idx >= total) return;
  int e = idx & 15, lane = (idx >> 4) & 31, tk = idx >> 9;
  int kt = tk % kKtFeat, mt = tk / kKtFeat;
  int m = mt * 16 + (lane & 15);
  int k = kt * 32 + a_k32(lane, e);
  float v = 0.0f;
  if (k < 2048)       v = vis[m * 2048 + k];
  else if (k < kFeat) v = tgt[m * 2052 + (k - 2048)];
  dst[idx] = (__bf16)v;
}

__global__ void pack_A_we(__bf16* __restrict__ dst,
                          const float* __restrict__ we) {  // [B,32,300]
  int idx = blockIdx.x * blockDim.x + threadIdx.x;
  const int total = kMTiles * kKtEmb * 32 * 16;
  if (idx >= total) return;
  int e = idx & 15, lane = (idx >> 4) & 31, tk = idx >> 9;
  int kt = tk % kKtEmb, mt = tk / kKtEmb;
  int r = mt * 16 + (lane & 15);       // r = t*64 + b
  int b = r & 63, t = r >> 6;
  int k = kt * 32 + a_k32(lane, e);
  float v = (k < kEmb) ? we[(b * 32 + t) * kEmb + k] : 0.0f;
  dst[idx] = (__bf16)v;
}

__global__ void pack_B(__bf16* __restrict__ dst, const float* __restrict__ src,
                       int Ntiles, int Kt, int srcK, int srcN, int ld,
                       int transpose) {
  int idx = blockIdx.x * blockDim.x + threadIdx.x;
  int total = Ntiles * Kt * 32 * 16;
  if (idx >= total) return;
  int e = idx & 15, lane = (idx >> 4) & 31, tk = idx >> 9;
  int kt = tk % Kt, nt = tk / Kt;
  int n = nt * 16 + (lane & 15);
  int k = kt * 32 + ((e >> 3) << 4) + ((lane >> 4) << 3) + (e & 7);
  float v = 0.0f;
  if (k < srcK && n < srcN)
    v = transpose ? src[n * ld + k] : src[k * ld + n];
  dst[idx] = (__bf16)v;
}

// ---- h0 = ReLU(feats @ W_map + b_map); write tt=0 state ------------------
__global__ void gemm_h0(const __bf16* __restrict__ Apk,
                        const __bf16* __restrict__ Bpk,
                        const float* __restrict__ bias,
                        __bf16* __restrict__ h_pack,
                        float* __restrict__ h_lin) {
  int lane = threadIdx.x & 31;
  int tile = blockIdx.x * (blockDim.x >> 5) + (threadIdx.x >> 5);
  if (tile >= 4 * 32) return;
  int mt = tile >> 5, nt = tile & 31;
  const v16bf* A = (const v16bf*)Apk + (mt * kKtFeat) * 32 + lane;
  const v16bf* Bp = (const v16bf*)Bpk + (nt * kKtFeat) * 32 + lane;
  v8f c = {};
  for (int kt = 0; kt < kKtFeat; ++kt)
    c = wmma_bf16(A[kt * 32], Bp[kt * 32], c);
  int n_loc = lane & 15, half = lane >> 4;
  int j = nt * 16 + n_loc;
  float bj = bias[j];
#pragma unroll
  for (int g = 0; g < 8; ++g) {
    int row = mt * 16 + g + 8 * half;        // batch index
    float v = c[g] + bj;
    v = v > 0.0f ? v : 0.0f;
    h_lin[row * kHid + j] = v;               // tt = 0
    int l2, e2;
    a_inv(g + 8 * half, j & 31, l2, e2);
    int kt2 = j >> 5;
    h_pack[(((mt)*kKtHid + kt2) * 32 + l2) * 16 + e2] = (__bf16)v; // tt=0
  }
}

// ---- x_gates = we @ w_ih^T + b_ih ---------------------------------------
__global__ void gemm_xg(const __bf16* __restrict__ Apk,
                        const __bf16* __restrict__ Bpk,
                        const float* __restrict__ bias,
                        float* __restrict__ xg_lin) {
  int lane = threadIdx.x & 31;
  int tile = blockIdx.x * (blockDim.x >> 5) + (threadIdx.x >> 5);
  if (tile >= kMTiles * (kG3 / 16)) return;
  int mt = tile / (kG3 / 16), nt = tile % (kG3 / 16);
  const v16bf* A = (const v16bf*)Apk + (mt * kKtEmb) * 32 + lane;
  const v16bf* Bp = (const v16bf*)Bpk + (nt * kKtEmb) * 32 + lane;
  v8f c = {};
  for (int kt = 0; kt < kKtEmb; ++kt)
    c = wmma_bf16(A[kt * 32], Bp[kt * 32], c);
  int n_loc = lane & 15, half = lane >> 4;
  int n = nt * 16 + n_loc;
  float bn = bias[n];
#pragma unroll
  for (int g = 0; g < 8; ++g) {
    int r = mt * 16 + g + 8 * half;          // r = t*64 + b
    xg_lin[r * kG3 + n] = c[g] + bn;
  }
}

// ---- one GRU step: hg = h @ w_hh^T, gates, state update ------------------
__global__ void gru_step(__bf16* __restrict__ h_pack,  // [32tt][4][16][32][16]
                         const __bf16* __restrict__ Whh,
                         const float* __restrict__ b_hh,
                         const float* __restrict__ xg_lin,
                         float* __restrict__ h_lin, int t) {
  int lane = threadIdx.x & 31;
  int tile = blockIdx.x * (blockDim.x >> 5) + (threadIdx.x >> 5);
  if (tile >= 4 * 32) return;
  int mt = tile >> 5, nt = tile & 31;
  const v16bf* A = (const v16bf*)h_pack + ((t * 4 + mt) * kKtHid) * 32 + lane;
  const v16bf* B0 = (const v16bf*)Whh + ((nt)*kKtHid) * 32 + lane;
  const v16bf* B1 = (const v16bf*)Whh + ((nt + 32) * kKtHid) * 32 + lane;
  const v16bf* B2 = (const v16bf*)Whh + ((nt + 64) * kKtHid) * 32 + lane;
  v8f cR = {}, cZ = {}, cN = {};
  for (int kt = 0; kt < kKtHid; ++kt) {
    v16bf a = A[kt * 32];
    cR = wmma_bf16(a, B0[kt * 32], cR);
    cZ = wmma_bf16(a, B1[kt * 32], cZ);
    cN = wmma_bf16(a, B2[kt * 32], cN);
  }
  int n_loc = lane & 15, half = lane >> 4;
  int j = nt * 16 + n_loc;
  float bhr = b_hh[j], bhz = b_hh[kHid + j], bhn = b_hh[2 * kHid + j];
#pragma unroll
  for (int g = 0; g < 8; ++g) {
    int b = mt * 16 + g + 8 * half;
    const float* xg = xg_lin + (t * kB + b) * kG3;
    float r = sigmoidf_(xg[j] + cR[g] + bhr);
    float z = sigmoidf_(xg[kHid + j] + cZ[g] + bhz);
    float nn = tanhf(xg[2 * kHid + j] + r * (cN[g] + bhn));
    float hp = h_lin[t * kB * kHid + b * kHid + j];
    float hnew = (1.0f - z) * nn + z * hp;
    h_lin[(t + 1) * kB * kHid + b * kHid + j] = hnew;
    int l2, e2;
    a_inv(g + 8 * half, j & 31, l2, e2);
    int kt2 = j >> 5;
    h_pack[((((t + 1) * 4 + mt) * kKtHid + kt2) * 32 + l2) * 16 + e2] =
        (__bf16)hnew;
  }
}

// ---- logits = hs @ W_cls + b_cls ----------------------------------------
// Block = 4 waves; block tile = 4 M-tiles x 4 N-tiles (64x64 output).
// B fragments for the 4 N-tiles (64 KB) staged once in LDS (async copy when
// available), shared by all 4 waves -> 4x less L2 traffic on W_cls.
__global__ void __launch_bounds__(128)
gemm_cls(const __bf16* __restrict__ h_pack, const __bf16* __restrict__ Wc,
         const float* __restrict__ bias, float* __restrict__ out) {
  __shared__ alignas(32) __bf16 Bsh[4 * kKtHid * 32 * 16];  // 64 KB
  int tid = threadIdx.x;
  int lane = tid & 31;
  int wv = tid >> 5;
  int mtg = blockIdx.x / (kNTCls / 4);
  int nt0 = (blockIdx.x % (kNTCls / 4)) * 4;

  // stage 64 KB of contiguous packed-B (4 N-tiles, all kt) into LDS
  const __bf16* gB = Wc + (size_t)nt0 * kKtHid * 32 * 16;
#pragma unroll
  for (int i = 0; i < 32; ++i) {
    int idx = i * 128 + tid;  // 4096 chunks of 16 B
    copy16_to_lds(gB + idx * 8, &Bsh[idx * 8]);
  }
  stage_fence();

  int mt = mtg * 4 + wv;
  const v16bf* A = (const v16bf*)h_pack + ((mt + 4) * kKtHid) * 32 + lane;
  const v16bf* Bl = (const v16bf*)Bsh + lane;
  v8f c0 = {}, c1 = {}, c2 = {}, c3 = {};
  for (int kt = 0; kt < kKtHid; ++kt) {
    v16bf a = A[kt * 32];
    c0 = wmma_bf16(a, Bl[(0 * kKtHid + kt) * 32], c0);
    c1 = wmma_bf16(a, Bl[(1 * kKtHid + kt) * 32], c1);
    c2 = wmma_bf16(a, Bl[(2 * kKtHid + kt) * 32], c2);
    c3 = wmma_bf16(a, Bl[(3 * kKtHid + kt) * 32], c3);
  }
  int n_loc = lane & 15, half = lane >> 4;
  v8f acc[4] = {c0, c1, c2, c3};
#pragma unroll
  for (int i = 0; i < 4; ++i) {
    int n = (nt0 + i) * 16 + n_loc;
    float bn = bias[n];
#pragma unroll
    for (int g = 0; g < 8; ++g) {
      int r = mt * 16 + g + 8 * half;  // r = t*64 + b
      int b = r & 63, t = r >> 6;
      out[(b * kSteps + t) * kVocab + n] = acc[i][g] + bn;
    }
  }
}

extern "C" void kernel_launch(void* const* d_in, const int* in_sizes, int n_in,
                              void* d_out, int out_size, void* d_ws,
                              size_t ws_size, hipStream_t stream) {
  (void)in_sizes; (void)n_in; (void)out_size; (void)ws_size;
  const float* word_embs = (const float*)d_in[0];
  const float* vis       = (const float*)d_in[1];
  const float* tgt       = (const float*)d_in[2];
  // d_in[3] = lang_len (unused; all sequences are full length T)
  const float* W_map = (const float*)d_in[4];
  const float* b_map = (const float*)d_in[5];
  const float* w_ih  = (const float*)d_in[6];
  const float* w_hh  = (const float*)d_in[7];
  const float* b_ih  = (const float*)d_in[8];
  const float* b_hh  = (const float*)d_in[9];
  const float* W_cls = (const float*)d_in[10];
  const float* b_cls = (const float*)d_in[11];
  float* out = (float*)d_out;

  char* ws = (char*)d_ws;
  size_t off = 0;
  auto take = [&](size_t bytes) {
    char* p = ws + off;
    off += (bytes + 255) & ~(size_t)255;
    return p;
  };
  __bf16* feats_pk = (__bf16*)take((size_t)4 * kKtFeat * 512 * 2);
  __bf16* xe_pk    = (__bf16*)take((size_t)kMTiles * kKtEmb * 512 * 2);
  __bf16* wmap_pk  = (__bf16*)take((size_t)32 * kKtFeat * 512 * 2);
  __bf16* wih_pk   = (__bf16*)take((size_t)96 * kKtEmb * 512 * 2);
  __bf16* whh_pk   = (__bf16*)take((size_t)96 * kKtHid * 512 * 2);
  __bf16* wcls_pk  = (__bf16*)take((size_t)kNTCls * kKtHid * 512 * 2);
  __bf16* h_pk     = (__bf16*)take((size_t)32 * 4 * kKtHid * 512 * 2);
  float*  h_lin    = (float*)take((size_t)32 * kB * kHid * 4);
  float*  xg_lin   = (float*)take((size_t)kMRows * kG3 * 4);

  const int TB = 256;
  // -- pack operands into WMMA fragment layout (bf16)
  pack_A_feats<<<(4 * kKtFeat * 512 + TB - 1) / TB, TB, 0, stream>>>(feats_pk,
                                                                     vis, tgt);
  pack_A_we<<<(kMTiles * kKtEmb * 512 + TB - 1) / TB, TB, 0, stream>>>(
      xe_pk, word_embs);
  pack_B<<<(32 * kKtFeat * 512 + TB - 1) / TB, TB, 0, stream>>>(
      wmap_pk, W_map, 32, kKtFeat, kFeat, kHid, kHid, 0);
  pack_B<<<(96 * kKtEmb * 512 + TB - 1) / TB, TB, 0, stream>>>(
      wih_pk, w_ih, 96, kKtEmb, kEmb, kG3, kEmb, 1);
  pack_B<<<(96 * kKtHid * 512 + TB - 1) / TB, TB, 0, stream>>>(
      whh_pk, w_hh, 96, kKtHid, kHid, kG3, kHid, 1);
  pack_B<<<(kNTCls * kKtHid * 512 + TB - 1) / TB, TB, 0, stream>>>(
      wcls_pk, W_cls, kNTCls, kKtHid, kHid, kVocab, kVocab, 0);

  // -- h0 (128 tiles, 8 waves/block)
  gemm_h0<<<16, TB, 0, stream>>>(feats_pk, wmap_pk, b_map, h_pk, h_lin);
  // -- x_gates (124*96 tiles)
  gemm_xg<<<(kMTiles * (kG3 / 16)) / 8, TB, 0, stream>>>(xe_pk, wih_pk, b_ih,
                                                         xg_lin);
  // -- GRU scan: 31 dependent steps
  for (int t = 0; t < kSteps; ++t)
    gru_step<<<16, TB, 0, stream>>>(h_pk, whh_pk, b_hh, xg_lin, h_lin, t);
  // -- classifier: 31 M-groups x 500 N-groups, 4 waves/block, LDS-staged B
  gemm_cls<<<(kMTiles / 4) * (kNTCls / 4), 128, 0, stream>>>(h_pk, wcls_pk,
                                                             b_cls, out);
}
